// GraphNeuralNetwork_23235773071650
// MI455X (gfx1250) — matrix operand order repaired and checked by reading, not compiled, not run on hardware
//
#include <hip/hip_runtime.h>

#define DD 512

typedef unsigned short u16;
typedef __attribute__((ext_vector_type(8)))  float  v8f_t;
typedef __attribute__((ext_vector_type(16))) __bf16 v16bf_t;

union Frag16 { v16bf_t v; uint4 q[2]; };

__device__ __forceinline__ u16 f2bf(float f) {
    union { float f; unsigned u; } c; c.f = f;
    unsigned r = c.u + 0x7FFFu + ((c.u >> 16) & 1u);   // round-to-nearest-even
    return (u16)(r >> 16);
}
__device__ __forceinline__ float bf2f(u16 b) {
    union { unsigned u; float f; } c; c.u = ((unsigned)b) << 16;
    return c.f;
}

// ---------------- degree / normalization ----------------
__global__ void k_init_deg(float* deg, int n) {
    int i = blockIdx.x * blockDim.x + threadIdx.x;
    if (i < n) deg[i] = 1.0f;                  // self-loop contribution
}
__global__ void k_accum_deg(const long long* __restrict__ dst, float* deg, int E) {
    int e = blockIdx.x * blockDim.x + threadIdx.x;
    if (e < E) unsafeAtomicAdd(&deg[(int)dst[e]], 1.0f);
}
__global__ void k_rsqrt(float* d, int n) {
    int i = blockIdx.x * blockDim.x + threadIdx.x;
    if (i < n) { float v = d[i]; d[i] = (v > 0.0f) ? rsqrtf(v) : 0.0f; }
}

// ---------------- fp32 -> bf16 (optional ReLU), 8 elems/thread ----------------
__global__ void k_cvt(const float* __restrict__ X, u16* __restrict__ Xb,
                      long long n8, int relu) {
    long long i = (long long)blockIdx.x * blockDim.x + threadIdx.x;
    if (i >= n8) return;
    const float4* p = (const float4*)X + i * 2;
    float4 a = p[0], b = p[1];
    if (relu) {
        a.x = fmaxf(a.x, 0.f); a.y = fmaxf(a.y, 0.f); a.z = fmaxf(a.z, 0.f); a.w = fmaxf(a.w, 0.f);
        b.x = fmaxf(b.x, 0.f); b.y = fmaxf(b.y, 0.f); b.z = fmaxf(b.z, 0.f); b.w = fmaxf(b.w, 0.f);
    }
    union { u16 s[8]; uint4 q; } o;
    o.s[0] = f2bf(a.x); o.s[1] = f2bf(a.y); o.s[2] = f2bf(a.z); o.s[3] = f2bf(a.w);
    o.s[4] = f2bf(b.x); o.s[5] = f2bf(b.y); o.s[6] = f2bf(b.z); o.s[7] = f2bf(b.w);
    ((uint4*)Xb)[i] = o.q;
}

// ---------------- pack W (row-major KxN fp32) into B-fragment layout ----------
// Tile (kb, nb): 32x16 bf16. Lane L holds column n = nb*16 + (L&15),
// K runs [koff..koff+7] and [koff+16..koff+23], koff = kb*32 + (L>=16 ? 8 : 0).
__global__ void k_pack_w(const float* __restrict__ W, u16* __restrict__ Wp) {
    int tile = blockIdx.x;            // 512 tiles = 16 kb * 32 nb
    int lane = threadIdx.x;           // 32 threads
    int kb = tile >> 5, nb = tile & 31;
    int n  = nb * 16 + (lane & 15);
    int k0 = kb * 32 + ((lane >> 4) ? 8 : 0);
    u16* dst = Wp + ((tile * 32 + lane) << 4);
#pragma unroll
    for (int i = 0; i < 16; ++i) dst[i] = f2bf(W[(k0 + (i & 7) + ((i >> 3) << 4)) * DD + n]);
}

// ---------------- bf16 WMMA GEMM: Out[N,512] = A[N,512] @ Wp ------------------
// Wave tile 32x64 (2 A-frags x 4 B-frags -> 8 WMMA / k-step), register
// double-buffered so next k-step's 12 loads overlap current 8 WMMAs.
__global__ __launch_bounds__(256) void k_gemm(const u16* __restrict__ A,
                                              const u16* __restrict__ Bp,
                                              u16* __restrict__ Out) {
    const int lane = threadIdx.x & 31;
    const int wid  = threadIdx.x >> 5;
    const int waveRow = wid & 3;          // 4 row-waves  -> 128 rows/block
    const int waveCol = wid >> 2;         // 2 col-waves  -> 128 cols/block
    const long long rowBase = (long long)blockIdx.y * 128 + waveRow * 32;
    const int colBase = blockIdx.x * 128 + waveCol * 64;
    const int m  = lane & 15;
    const int hi = lane >> 4;
    // A fragment (16-bit 16x32 layout): lane = row m, koff = hi*8,
    // needs K runs [koff..koff+7] and [koff+16..koff+23].
    const u16* aRow0 = A + (rowBase + m) * DD + hi * 8;
    const u16* aRow1 = aRow0 + 16 * DD;
    const int  nbBase = colBase >> 4;
    const u16* bBase = Bp + ((nbBase * 32 + lane) << 4);   // +t*512/elem per col tile

    v8f_t  acc[8] = {};
    Frag16 aF[2][2], bF[2][4];

    // preload k-step 0
    aF[0][0].q[0] = *(const uint4*)(aRow0);
    aF[0][0].q[1] = *(const uint4*)(aRow0 + 16);
    aF[0][1].q[0] = *(const uint4*)(aRow1);
    aF[0][1].q[1] = *(const uint4*)(aRow1 + 16);
#pragma unroll
    for (int t = 0; t < 4; ++t) {
        bF[0][t].q[0] = *(const uint4*)(bBase + t * 512);
        bF[0][t].q[1] = *(const uint4*)(bBase + t * 512 + 8);
    }

#pragma unroll
    for (int kb = 0; kb < DD / 32; ++kb) {
        const int cur = kb & 1, nxt = cur ^ 1;
        if (kb + 1 < DD / 32) {            // issue next k-step's loads first
            const u16* a0 = aRow0 + (kb + 1) * 32;
            const u16* a1 = aRow1 + (kb + 1) * 32;
            const u16* b  = bBase + (kb + 1) * 16384;   // 32 tiles * 32 lanes * 16
            aF[nxt][0].q[0] = *(const uint4*)(a0);
            aF[nxt][0].q[1] = *(const uint4*)(a0 + 16);
            aF[nxt][1].q[0] = *(const uint4*)(a1);
            aF[nxt][1].q[1] = *(const uint4*)(a1 + 16);
#pragma unroll
            for (int t = 0; t < 4; ++t) {
                bF[nxt][t].q[0] = *(const uint4*)(b + t * 512);
                bF[nxt][t].q[1] = *(const uint4*)(b + t * 512 + 8);
            }
        }
#pragma unroll
        for (int t = 0; t < 4; ++t) {
            acc[t]     = __builtin_amdgcn_wmma_f32_16x16x32_bf16(
                false, aF[cur][0].v, false, bF[cur][t].v, (short)0, acc[t],     false, false);
            acc[4 + t] = __builtin_amdgcn_wmma_f32_16x16x32_bf16(
                false, aF[cur][1].v, false, bF[cur][t].v, (short)0, acc[4 + t], false, false);
        }
    }

    // C/D layout: column = lane&15, row = vgpr + 8*hi
#pragma unroll
    for (int rt = 0; rt < 2; ++rt) {
        long long r0 = rowBase + rt * 16 + hi * 8;
#pragma unroll
        for (int t = 0; t < 4; ++t) {
            int c = colBase + t * 16 + m;
#pragma unroll
            for (int v = 0; v < 8; ++v)
                Out[(r0 + v) * DD + c] = f2bf(acc[rt * 4 + t][v]);
        }
    }
}

// ---------------- Out[i,j] = bias[j] + dinv[i]^2 * H[i,j]  (self loop) --------
__global__ void k_self_init(const float* __restrict__ bias, const float* __restrict__ dinv,
                            const u16* __restrict__ Hb, float* __restrict__ Out, int N) {
    int t = blockIdx.x * blockDim.x + threadIdx.x;
    int row = t >> 7;                  // 128 threads per row (512/4)
    if (row >= N) return;
    int j = (t & 127) << 2;
    float w = dinv[row]; w = w * w;
    unsigned long long hv = *(const unsigned long long*)(Hb + (long long)row * DD + j);
    float4 bv = *(const float4*)(bias + j);
    float4 o;
    o.x = bv.x + w * bf2f((u16)hv);
    o.y = bv.y + w * bf2f((u16)(hv >> 16));
    o.z = bv.z + w * bf2f((u16)(hv >> 32));
    o.w = bv.w + w * bf2f((u16)(hv >> 48));
    *(float4*)(Out + (long long)row * DD + j) = o;
}

// ---------------- edge scatter: Out[dst] += norm * H[src] ---------------------
__global__ void k_scatter(const long long* __restrict__ src, const long long* __restrict__ dst,
                          const float* __restrict__ dinv, const u16* __restrict__ Hb,
                          float* __restrict__ Out, int E) {
    int e = blockIdx.x * 2 + (threadIdx.x >> 7);   // 2 edges per 256-thread block
    if (e >= E) return;
    int j = (threadIdx.x & 127) << 2;
    long long s = src[e], d = dst[e];
    float norm = dinv[s] * dinv[d];
    unsigned long long hv = *(const unsigned long long*)(Hb + s * DD + j);
    float* o = Out + d * DD + j;
    unsafeAtomicAdd(o + 0, norm * bf2f((u16)hv));
    unsafeAtomicAdd(o + 1, norm * bf2f((u16)(hv >> 16)));
    unsafeAtomicAdd(o + 2, norm * bf2f((u16)(hv >> 32)));
    unsafeAtomicAdd(o + 3, norm * bf2f((u16)(hv >> 48)));
}

extern "C" void kernel_launch(void* const* d_in, const int* in_sizes, int n_in,
                              void* d_out, int out_size, void* d_ws, size_t ws_size,
                              hipStream_t stream) {
    const float*     x   = (const float*)d_in[0];
    const long long* ei  = (const long long*)d_in[1];
    const float*     W1  = (const float*)d_in[2];
    const float*     b1  = (const float*)d_in[3];
    const float*     W2  = (const float*)d_in[4];
    const float*     b2  = (const float*)d_in[5];
    const int N = in_sizes[0] / DD;        // 65536
    const int E = in_sizes[1] / 2;         // 524288
    const long long* srcI = ei;            // edge_index row 0
    const long long* dstI = ei + E;        // edge_index row 1

    char* ws = (char*)d_ws;
    float* dinv = (float*)ws;                                       // N*4
    u16*   W1p  = (u16*)(ws + (1ull << 20));                        // 512 KB
    u16*   W2p  = (u16*)(ws + (1ull << 20) + (512ull << 10));       // 512 KB
    u16*   Xb   = (u16*)(ws + (2ull << 20));                        // 64 MB
    u16*   Hb   = (u16*)(ws + (2ull << 20) + ((long long)N * DD * 2)); // 64 MB
    float* A1   = (float*)d_out;           // fp32 scatter accumulator, reused

    const long long n8 = (long long)N * DD / 8;

    // normalization: deg (with self loop) -> deg^{-1/2}
    k_init_deg <<<(N + 255) / 256, 256, 0, stream>>>(dinv, N);
    k_accum_deg<<<(E + 255) / 256, 256, 0, stream>>>(dstI, dinv, E);
    k_rsqrt    <<<(N + 255) / 256, 256, 0, stream>>>(dinv, N);

    // prep: X -> bf16, pack weights into B-fragment layout
    k_cvt   <<<(unsigned)((n8 + 255) / 256), 256, 0, stream>>>(x, Xb, n8, 0);
    k_pack_w<<<512, 32, 0, stream>>>(W1, W1p);
    k_pack_w<<<512, 32, 0, stream>>>(W2, W2p);

    dim3 gemmGrid(DD / 128, N / 128);

    // ---- layer 1 ----
    k_gemm     <<<gemmGrid, 256, 0, stream>>>(Xb, W1p, Hb);                 // H = X @ W1 (bf16)
    k_self_init<<<N / 2, 256, 0, stream>>>(b1, dinv, Hb, A1, N);            // b1 + self loop
    k_scatter  <<<(E + 1) / 2, 256, 0, stream>>>(srcI, dstI, dinv, Hb, A1, E);
    k_cvt      <<<(unsigned)((n8 + 255) / 256), 256, 0, stream>>>(A1, Xb, n8, 1); // ReLU -> bf16

    // ---- layer 2 ----
    k_gemm     <<<gemmGrid, 256, 0, stream>>>(Xb, W2p, Hb);                 // H = relu(A1) @ W2
    k_self_init<<<N / 2, 256, 0, stream>>>(b2, dinv, Hb, (float*)d_out, N); // b2 + self loop
    k_scatter  <<<(E + 1) / 2, 256, 0, stream>>>(srcI, dstI, dinv, Hb, (float*)d_out, E);
}
